// SwinTransformerBlock_26749056319838
// MI455X (gfx1250) — matrix operand order repaired and checked by reading, not compiled, not run on hardware
//
#include <hip/hip_runtime.h>

// ---------------- CDNA5 (gfx1250) wave32 WMMA bf16 Swin block ----------------
typedef __attribute__((ext_vector_type(16))) __bf16 v16bf;
typedef __attribute__((ext_vector_type(8)))  __bf16 v8bf;
typedef __attribute__((ext_vector_type(4)))  __bf16 v4bf;
typedef __attribute__((ext_vector_type(8)))  float  v8f;

#define DIM   256
#define Hh    56
#define Ww    56
#define WS    7
#define SHIFT 3
#define NHEAD 8
#define HD    32
#define NTOK  49          // tokens per window
#define NWIN  64          // windows per image
#define BATCH 32
#define MROWS (BATCH * Hh * Ww)   // 100352 token rows
#define SCALE 0.17677669529663687f

__device__ __forceinline__ v8f wmma_bf16(v16bf a, v16bf b, v8f c) {
  return __builtin_amdgcn_wmma_f32_16x16x32_bf16(false, a, false, b, (short)0, c,
                                                 false, false);
}

// A fragment (16x32 bf16, row-major source)
__device__ __forceinline__ v16bf frag_a(const __bf16* base, int stride,
                                        int row_base, int k0, int lane) {
  int half = (lane >> 4) & 1, l = lane & 15;
  const __bf16* p = base + (size_t)(row_base + l) * stride + k0;
  v8bf lo = *(const v8bf*)(p + half * 8);
  v8bf hi = *(const v8bf*)(p + 16 + half * 8);
  return __builtin_shufflevector(lo, hi, 0,1,2,3,4,5,6,7,8,9,10,11,12,13,14,15);
}

// B fragment (32x16 bf16): row index = output column n, contiguous K in row.
__device__ __forceinline__ v16bf frag_b(const __bf16* base, int stride,
                                        int col_base, int k0, int lane) {
  int half = (lane >> 4) & 1, l = lane & 15;
  const __bf16* p = base + (size_t)(col_base + l) * stride + k0;
  v8bf lo = *(const v8bf*)(p + half * 8);
  v8bf hi = *(const v8bf*)(p + 16 + half * 8);
  return __builtin_shufflevector(lo, hi, 0,1,2,3,4,5,6,7,8,9,10,11,12,13,14,15);
}

// ---------------------------------------------------------------- fp32 -> bf16
__global__ void cvt_kernel(const float* __restrict__ s, __bf16* __restrict__ d,
                           int n) {
  for (int i = blockIdx.x * blockDim.x + threadIdx.x; i < n;
       i += gridDim.x * blockDim.x)
    d[i] = (__bf16)s[i];
}

// ----------------------- rel-position bias table in WMMA C-fragment order ----
__global__ void bias_kernel(const float* __restrict__ rel,
                            float* __restrict__ biasC) {
  int idx = blockIdx.x * 256 + threadIdx.x;   // 32768 total
  int r = idx & 7, lane = (idx >> 3) & 31, nt = (idx >> 8) & 3;
  int s = (idx >> 10) & 3, h = idx >> 12;
  int m = s * 16 + r + (lane >> 4) * 8;
  int c = nt * 16 + (lane & 15);
  float v = -1.0e30f;
  if (m < NTOK && c < NTOK) {
    int ridx = (m / WS - c / WS + (WS - 1)) * (2 * WS - 1) +
               (m % WS - c % WS + (WS - 1));
    v = rel[ridx * NHEAD + h];
  }
  biasC[idx] = v;
}

// ------------------------------------------- LayerNorm (+optional shift/window)
__global__ __launch_bounds__(256)
void ln_kernel(const float* __restrict__ src, const float* __restrict__ g,
               const float* __restrict__ bta, __bf16* __restrict__ dst,
               int mode) {
  int wave = threadIdx.x >> 5, lane = threadIdx.x & 31;
  int r = blockIdx.x * 8 + wave;
  int srow = r;
  if (mode == 1) {
    int b = r / (Hh * Ww), t = r % (Hh * Ww);
    int w = t / NTOK, n = t % NTOK;
    int wh = w >> 3, ww = w & 7, ii = n / WS, jj = n % WS;
    int hs = (wh * WS + ii + SHIFT) % Hh;
    int ws = (ww * WS + jj + SHIFT) % Ww;
    srow = b * (Hh * Ww) + hs * Ww + ws;
  }
  const float* p = src + (size_t)srow * DIM + lane * 8;
  float4 a0 = *(const float4*)p;
  float4 a1 = *(const float4*)(p + 4);
  float v[8] = {a0.x, a0.y, a0.z, a0.w, a1.x, a1.y, a1.z, a1.w};
  float s = 0.f, q = 0.f;
#pragma unroll
  for (int k = 0; k < 8; k++) { s += v[k]; q += v[k] * v[k]; }
#pragma unroll
  for (int o = 16; o > 0; o >>= 1) {
    s += __shfl_xor(s, o, 32);
    q += __shfl_xor(q, o, 32);
  }
  float mean = s * (1.f / DIM);
  float var  = q * (1.f / DIM) - mean * mean;
  float rstd = rsqrtf(var + 1e-5f);
  int ch = lane * 8;
  __attribute__((aligned(16))) __bf16 ov[8];
#pragma unroll
  for (int k = 0; k < 8; k++)
    ov[k] = (__bf16)((v[k] - mean) * rstd * g[ch + k] + bta[ch + k]);
  *(v8bf*)(dst + (size_t)r * DIM + ch) = *(const v8bf*)ov;
}

// ------------------------------------------------------------------ WMMA GEMM
// Block tile 128x128, 8 waves 4(M)x2(N), wave tile 32x64, double-buffered LDS.
// EPI: 0 = QKV (bf16, q-cols pre-scaled), 1 = proj+reverse-shift+residual,
//      2 = exact GELU -> bf16, 3 = residual add -> fp32.
template <int EPI>
__global__ __launch_bounds__(256)
void gemm_kernel(const __bf16* __restrict__ A, const __bf16* __restrict__ W,
                 const float* __restrict__ bias, int M, int Nc, int K,
                 const float* __restrict__ resin, float* __restrict__ outf,
                 __bf16* __restrict__ outb) {
  __shared__ __bf16 As[2][128 * 32];
  __shared__ __bf16 Wsh[2][128 * 32];
  int tid = threadIdx.x, lane = tid & 31, wave = tid >> 5;
  int wm = wave >> 1, wn = wave & 1;
  int mBase = blockIdx.y * 128, nBase = blockIdx.x * 128;

  v8f zero = {};
  v8f acc[2][4];
#pragma unroll
  for (int i = 0; i < 2; i++)
#pragma unroll
    for (int j = 0; j < 4; j++) acc[i][j] = zero;

  int arow = tid >> 1, aseg = tid & 1;   // 128 rows x 2 16-elem segs
  const __bf16* aptr = A + (size_t)(mBase + arow) * K + aseg * 16;
  const __bf16* wptr = W + (size_t)(nBase + arow) * K + aseg * 16;
  int lofs = arow * 32 + aseg * 16;

  int nslab = K >> 5;
  {
    v8bf a0 = *(const v8bf*)aptr;
    v8bf a1 = *(const v8bf*)(aptr + 8);
    v8bf w0 = *(const v8bf*)wptr;
    v8bf w1 = *(const v8bf*)(wptr + 8);
    *(v8bf*)(&As[0][lofs])      = a0;
    *(v8bf*)(&As[0][lofs + 8])  = a1;
    *(v8bf*)(&Wsh[0][lofs])     = w0;
    *(v8bf*)(&Wsh[0][lofs + 8]) = w1;
  }
  __syncthreads();

  for (int kk = 0; kk < nslab; kk++) {
    int cur = kk & 1;
    if (kk + 1 < nslab) {
      int nxt = cur ^ 1;
      const __bf16* ap = aptr + (kk + 1) * 32;
      const __bf16* wp = wptr + (kk + 1) * 32;
      v8bf a0 = *(const v8bf*)ap;
      v8bf a1 = *(const v8bf*)(ap + 8);
      v8bf w0 = *(const v8bf*)wp;
      v8bf w1 = *(const v8bf*)(wp + 8);
      if (kk + 2 < nslab) {               // global_prefetch two slabs ahead
        __builtin_prefetch(ap + 32, 0, 1);
        __builtin_prefetch(wp + 32, 0, 1);
      }
      *(v8bf*)(&As[nxt][lofs])      = a0;
      *(v8bf*)(&As[nxt][lofs + 8])  = a1;
      *(v8bf*)(&Wsh[nxt][lofs])     = w0;
      *(v8bf*)(&Wsh[nxt][lofs + 8]) = w1;
    }
    v16bf af0 = frag_a(&As[cur][0], 32, wm * 32,      0, lane);
    v16bf af1 = frag_a(&As[cur][0], 32, wm * 32 + 16, 0, lane);
    v16bf bf[4];
#pragma unroll
    for (int j = 0; j < 4; j++)
      bf[j] = frag_b(&Wsh[cur][0], 32, wn * 64 + j * 16, 0, lane);
#pragma unroll
    for (int j = 0; j < 4; j++) {
      acc[0][j] = wmma_bf16(af0, bf[j], acc[0][j]);
      acc[1][j] = wmma_bf16(af1, bf[j], acc[1][j]);
    }
    __syncthreads();
  }

  int half = lane >> 4, lcol = lane & 15;
#pragma unroll
  for (int i = 0; i < 2; i++)
#pragma unroll
    for (int j = 0; j < 4; j++) {
      int col = nBase + wn * 64 + j * 16 + lcol;
      float bc = bias[col];
#pragma unroll
      for (int r = 0; r < 8; r++) {
        int row = mBase + wm * 32 + i * 16 + r + half * 8;
        float v = acc[i][j][r] + bc;
        if (EPI == 0) {
          if (col < DIM) v *= SCALE;      // pre-scale Q columns
          outb[(size_t)row * Nc + col] = (__bf16)v;
        } else if (EPI == 2) {
          float gel = 0.5f * v * (1.f + erff(v * 0.70710678118f));
          outb[(size_t)row * Nc + col] = (__bf16)gel;
        } else if (EPI == 3) {
          outf[(size_t)row * Nc + col] = resin[(size_t)row * Nc + col] + v;
        } else {  // EPI == 1 : window-major row -> natural row (reverse shift)
          int b = row / (Hh * Ww), t = row % (Hh * Ww);
          int w = t / NTOK, n = t % NTOK;
          int wh = w >> 3, ww = w & 7, ii = n / WS, jj = n % WS;
          int dh = (wh * WS + ii + SHIFT) % Hh;
          int dw = (ww * WS + jj + SHIFT) % Ww;
          size_t nrow = (size_t)b * (Hh * Ww) + dh * Ww + dw;
          outf[nrow * DIM + col] = resin[nrow * DIM + col] + v;
        }
      }
    }
}

// --------------------------------------------------------------- attention
// One block per window (2048), one wave per head.  Q/K fragments from global;
// V transposed into LDS (token axis stored PERMUTED: pos = lcol*4 + nt, so
// each lane's 4 prob values pack into one ds_store_b64) with an appended
// ones-row so softmax row sums fall out of an extra WMMA N-tile.
#define ATTN_REG_OFF   0
#define ATTN_VT_OFF    256
#define ATTN_PS_OFF    (256 + 8 * 3072 * 2)           // 49408
#define ATTN_SMEM_BYTES (ATTN_PS_OFF + 8 * 1024 * 2)  // 65792

__global__ __launch_bounds__(256)
void attn_kernel(const __bf16* __restrict__ qkv, const float* __restrict__ biasC,
                 __bf16* __restrict__ out) {
  extern __shared__ char smem[];
  int*    reg   = (int*)(smem + ATTN_REG_OFF);
  __bf16* VtAll = (__bf16*)(smem + ATTN_VT_OFF);   // per head 48x64 (permuted)
  __bf16* PsAll = (__bf16*)(smem + ATTN_PS_OFF);   // per wave 16x64 (permuted)

  int tid = threadIdx.x, lane = tid & 31, wave = tid >> 5;
  int win = blockIdx.x;
  int wrow = win * NTOK;
  int wloc = win & (NWIN - 1);
  bool boundary = ((wloc >> 3) == 7) || ((wloc & 7) == 7);

  if (tid < 64) {
    int t = tid, rg = -1;
    if (t < NTOK) {
      int ii = t / WS, jj = t % WS;
      int hh = (wloc >> 3) * WS + ii, ww = (wloc & 7) * WS + jj;
      int rr = hh < 49 ? 0 : (hh < 53 ? 1 : 2);
      int cc = ww < 49 ? 0 : (ww < 53 ? 1 : 2);
      rg = rr * 3 + cc;
    }
    reg[t] = rg;
  }

  int h = wave;
  __bf16* Vt = VtAll + h * 3072;     // Vt[hd][perm]; hd 32 = ones, 33..47 = 0
  __bf16* Ps = PsAll + wave * 1024;  // 16 x 64 probs stripe (perm columns)
  const __bf16* qbase = qkv + (size_t)wrow * (3 * DIM) + h * HD;
  const __bf16* kbase = qbase + DIM;

  for (int t = lane; t < 64; t += 32) {
    int p = (t & 15) * 4 + (t >> 4);          // permuted token position
    __bf16 z = (__bf16)0.0f;
    if (t < NTOK) {
      const v8bf* vv = (const v8bf*)(qkv + (size_t)(wrow + t) * (3 * DIM) +
                                     2 * DIM + h * HD);
      __attribute__((aligned(16))) __bf16 vb[32];
#pragma unroll
      for (int c = 0; c < 4; c++) *(v8bf*)(vb + c * 8) = vv[c];
#pragma unroll
      for (int d = 0; d < 32; d++) Vt[d * 64 + p] = vb[d];
    } else {
#pragma unroll
      for (int d = 0; d < 32; d++) Vt[d * 64 + p] = z;
    }
    Vt[32 * 64 + p] = (__bf16)1.0f;          // ones row -> row sums via WMMA
#pragma unroll
    for (int d = 33; d < 48; d++) Vt[d * 64 + p] = z;
  }
  __syncthreads();

  int half = lane >> 4, lcol = lane & 15;
  int creg[4];
  if (boundary) {
#pragma unroll
    for (int nt = 0; nt < 4; nt++) creg[nt] = reg[nt * 16 + lcol];
  }

  for (int s = 0; s < 4; s++) {
    // ---- scores: S = Qscaled(16x32) @ K^T + biasC (WMMA C operand) ----
    v16bf aq  = frag_a(qbase, 3 * DIM, s * 16, 0, lane);
    v16bf bk0 = frag_b(kbase, 3 * DIM, 0,  0, lane);
    v16bf bk1 = frag_b(kbase, 3 * DIM, 16, 0, lane);
    v16bf bk2 = frag_b(kbase, 3 * DIM, 32, 0, lane);
    v16bf bk3 = frag_b(kbase, 3 * DIM, 48, 0, lane);
    const float* bC = biasC + (size_t)((h * 4 + s) * 4) * 256 + lane * 8;
    v8f cf[4];
#pragma unroll
    for (int nt = 0; nt < 4; nt++) cf[nt] = *(const v8f*)(bC + nt * 256);
    if (boundary) {
      int mreg[8];
#pragma unroll
      for (int r = 0; r < 8; r++) mreg[r] = reg[s * 16 + r + half * 8];
#pragma unroll
      for (int nt = 0; nt < 4; nt++)
#pragma unroll
        for (int r = 0; r < 8; r++)
          cf[nt][r] += (mreg[r] == creg[nt]) ? 0.f : -100.f;
    }
    v8f sc[4];
    sc[0] = wmma_bf16(aq, bk0, cf[0]);
    sc[1] = wmma_bf16(aq, bk1, cf[1]);
    sc[2] = wmma_bf16(aq, bk2, cf[2]);
    sc[3] = wmma_bf16(aq, bk3, cf[3]);

    // ---- exp -> packed b64 stores into permuted P stripe ----
#pragma unroll
    for (int r = 0; r < 8; r++) {
      __attribute__((aligned(8))) __bf16 pk[4];
#pragma unroll
      for (int nt = 0; nt < 4; nt++) pk[nt] = (__bf16)__expf(sc[nt][r]);
      *(v4bf*)(Ps + (r + half * 8) * 64 + lcol * 4) = *(const v4bf*)pk;
    }

    // ---- [O | rowsum] = P(16x64) @ [V | 1](64x48), permuted K axis ----
    v8f zero = {};
    v8f oacc0 = zero, oacc1 = zero, osum = zero;
#pragma unroll
    for (int kt = 0; kt < 2; kt++) {
      v16bf pa  = frag_a(Ps, 64, 0, kt * 32, lane);
      v16bf vb0 = frag_b(Vt, 64, 0,  kt * 32, lane);
      v16bf vb1 = frag_b(Vt, 64, 16, kt * 32, lane);
      v16bf vbS = frag_b(Vt, 64, 32, kt * 32, lane);
      oacc0 = wmma_bf16(pa, vb0, oacc0);
      oacc1 = wmma_bf16(pa, vb1, oacc1);
      osum  = wmma_bf16(pa, vbS, osum);
    }
    // row sums live in tile col 0 (lanes 0 / 16); broadcast within half-wave
    float rinv[8];
#pragma unroll
    for (int r = 0; r < 8; r++)
      rinv[r] = __builtin_amdgcn_rcpf(__shfl(osum[r], 0, 16));
#pragma unroll
    for (int r = 0; r < 8; r++) {
      int m = s * 16 + r + half * 8;
      if (m < NTOK) {
        __bf16* op = out + (size_t)(wrow + m) * DIM + h * HD + lcol;
        op[0]  = (__bf16)(oacc0[r] * rinv[r]);
        op[16] = (__bf16)(oacc1[r] * rinv[r]);
      }
    }
  }
}

// ------------------------------------------------------------------- launch
extern "C" void kernel_launch(void* const* d_in, const int* in_sizes, int n_in,
                              void* d_out, int out_size, void* d_ws,
                              size_t ws_size, hipStream_t stream) {
  const float* x       = (const float*)d_in[0];
  const float* n1g     = (const float*)d_in[1];
  const float* n1b     = (const float*)d_in[2];
  const float* qkv_w   = (const float*)d_in[3];
  const float* qkv_b   = (const float*)d_in[4];
  const float* rel_tab = (const float*)d_in[5];
  const float* proj_w  = (const float*)d_in[6];
  const float* proj_b  = (const float*)d_in[7];
  const float* n2g     = (const float*)d_in[8];
  const float* n2b     = (const float*)d_in[9];
  const float* fc1_w   = (const float*)d_in[10];
  const float* fc1_b   = (const float*)d_in[11];
  const float* fc2_w   = (const float*)d_in[12];
  const float* fc2_b   = (const float*)d_in[13];
  float* outp = (float*)d_out;

  char* ws = (char*)d_ws;
  __bf16* wq    = (__bf16*)(ws + 0);           // 768*256
  __bf16* wp    = (__bf16*)(ws + 393216);      // 256*256
  __bf16* w1    = (__bf16*)(ws + 524288);      // 1024*256
  __bf16* w2    = (__bf16*)(ws + 1048576);     // 256*1024
  float*  biasC = (float*)(ws + 1572864);      // 8*4*4*256 f32 (128KB)
  __bf16* xw    = (__bf16*)(ws + 1703936);     // 100352*256 bf16
  __bf16* qkvb  = (__bf16*)(ws + 53084160);    // 100352*768 bf16; reused as h1
  __bf16* h1    = qkvb;
  __bf16* attn  = (__bf16*)(ws + 258605056);   // 100352*256 bf16; reused as y2
  __bf16* y2    = attn;
  float*  xres  = (float*)(ws + 309985280);    // 100352*256 fp32

  cvt_kernel<<<256, 256, 0, stream>>>(qkv_w, wq, 768 * 256);
  cvt_kernel<<<256, 256, 0, stream>>>(proj_w, wp, 256 * 256);
  cvt_kernel<<<256, 256, 0, stream>>>(fc1_w, w1, 1024 * 256);
  cvt_kernel<<<256, 256, 0, stream>>>(fc2_w, w2, 256 * 1024);
  bias_kernel<<<128, 256, 0, stream>>>(rel_tab, biasC);

  ln_kernel<<<MROWS / 8, 256, 0, stream>>>(x, n1g, n1b, xw, 1);

  gemm_kernel<0><<<dim3(768 / 128, MROWS / 128), 256, 0, stream>>>(
      xw, wq, qkv_b, MROWS, 768, 256, nullptr, nullptr, qkvb);

  (void)hipFuncSetAttribute((const void*)attn_kernel,
                            hipFuncAttributeMaxDynamicSharedMemorySize,
                            ATTN_SMEM_BYTES);
  attn_kernel<<<BATCH * NWIN, 256, ATTN_SMEM_BYTES, stream>>>(qkvb, biasC,
                                                              attn);

  gemm_kernel<1><<<dim3(256 / 128, MROWS / 128), 256, 0, stream>>>(
      attn, wp, proj_b, MROWS, 256, 256, x, xres, nullptr);

  ln_kernel<<<MROWS / 8, 256, 0, stream>>>(xres, n2g, n2b, y2, 0);

  gemm_kernel<2><<<dim3(1024 / 128, MROWS / 128), 256, 0, stream>>>(
      y2, w1, fc1_b, MROWS, 1024, 256, nullptr, nullptr, h1);

  gemm_kernel<3><<<dim3(256 / 128, MROWS / 128), 256, 0, stream>>>(
      h1, w2, fc2_b, MROWS, 256, 1024, xres, outp, nullptr);
}
